// UGCABlock_77429670412500
// MI455X (gfx1250) — compile-verified, matrix operand order checked by reading, not run on hardware
//
#include <hip/hip_runtime.h>
#include <hip/hip_bf16.h>
#include <math.h>

// ---------------- CDNA5 WMMA plumbing (gfx1250, wave32) ----------------
typedef __bf16 bf16_t;
typedef __attribute__((ext_vector_type(16))) __bf16 v16bf;
typedef __attribute__((ext_vector_type(8)))  __bf16 v8bf;
typedef __attribute__((ext_vector_type(8)))  float  v8f;

union frag_u { v16bf v; v8bf h[2]; };

#define D_MODEL 768
#define NHEAD   12
#define DHEAD   64
#define DFF     3072
#define LN_EPS  1e-5f
#define BATCH   16
#define LQ_D    512
#define LQ_P    1024

__device__ __forceinline__ v8f wmma_bf16(v16bf a, v16bf b, v8f c) {
  // D = A(16x32 bf16) x B(32x16 bf16) + C(16x16 f32)
  return __builtin_amdgcn_wmma_f32_16x16x32_bf16(
      /*neg_a=*/false, a, /*neg_b=*/false, b,
      /*c_mod=*/(short)0, c, /*reuse_a=*/false, /*reuse_b=*/false);
}

// Packed LDS fragment load: per-lane two contiguous 16B runs (kb and kb+16 elems).
__device__ __forceinline__ v16bf load_frag_lds(const bf16_t* rowbase, int kb) {
  frag_u u;
  u.h[0] = *(const v8bf*)(rowbase + kb);
  u.h[1] = *(const v8bf*)(rowbase + kb + 16);
  return u.v;
}

// A fragment (16x32 bf16) from row-major f32 global: lanes 0-15 row=lane,
// K {0..7,16..23}; lanes 16-31 row=lane-16, K {8..15,24..31}. float4 loads.
__device__ __forceinline__ v16bf load_a_frag_g(const float* __restrict__ p, int ld) {
  const int lane = threadIdx.x & 31;
  const int row  = lane & 15;
  const int kb   = (lane < 16) ? 0 : 8;
  const float* r = p + (size_t)row * ld + kb;
  const float4 x0 = *(const float4*)(r);
  const float4 x1 = *(const float4*)(r + 4);
  const float4 x2 = *(const float4*)(r + 16);
  const float4 x3 = *(const float4*)(r + 20);
  v16bf a;
  a[0]=(bf16_t)x0.x;  a[1]=(bf16_t)x0.y;  a[2]=(bf16_t)x0.z;  a[3]=(bf16_t)x0.w;
  a[4]=(bf16_t)x1.x;  a[5]=(bf16_t)x1.y;  a[6]=(bf16_t)x1.z;  a[7]=(bf16_t)x1.w;
  a[8]=(bf16_t)x2.x;  a[9]=(bf16_t)x2.y;  a[10]=(bf16_t)x2.z; a[11]=(bf16_t)x2.w;
  a[12]=(bf16_t)x3.x; a[13]=(bf16_t)x3.y; a[14]=(bf16_t)x3.z; a[15]=(bf16_t)x3.w;
  return a;
}

// B fragment (32x16 bf16): lane holds column n=lane&15, element (k,n)=p[k*ld+n].
__device__ __forceinline__ v16bf load_b_frag_g(const float* __restrict__ p, int ld) {
  const int lane = threadIdx.x & 31;
  const int col  = lane & 15;
  const int kb   = (lane < 16) ? 0 : 8;
  v16bf b;
#pragma unroll
  for (int i = 0; i < 8; ++i) b[i]     = (bf16_t)p[(size_t)(kb + i) * ld + col];
#pragma unroll
  for (int i = 0; i < 8; ++i) b[8 + i] = (bf16_t)p[(size_t)(kb + 16 + i) * ld + col];
  return b;
}

__device__ __forceinline__ float gelu_exact(float x) {
  return 0.5f * x * (1.0f + erff(x * 0.70710678118654752f));
}
__device__ __forceinline__ float softplus_f(float x) {
  return (x > 20.f) ? x : log1pf(expf(x));
}

// ---------------- WMMA GEMM: C = act(A[M,K] @ W[K,N] + bias) ----------------
// 128x64 block tile, 8 waves (one 16-row tile each, 4 col-tiles -> 4 WMMA/chunk).
// W tile staged TRANSPOSED in LDS so B fragments are contiguous b128 loads.
#define BM 128
#define BN 64
#define BK 32
#define LSTR 40  // bf16 row stride (80B, multiple of 16B)

__global__ __launch_bounds__(256) void gemm_bias_act_kernel(
    const float* __restrict__ A, const float* __restrict__ W,
    const float* __restrict__ bias, float* __restrict__ C,
    int M, int N, int K, int act) {
  __shared__ bf16_t As[BM * LSTR];  // [row][k]
  __shared__ bf16_t Wt[BN * LSTR];  // [n][k]  (transposed W tile)

  const int tid  = threadIdx.x;
  const int wid  = tid >> 5;
  const int lane = tid & 31;
  const int m0 = blockIdx.y * BM;
  const int n0 = blockIdx.x * BN;

  // staging indices
  const int ar = tid >> 1;            // 0..127
  const int ac = (tid & 1) * 16;      // 0,16
  const int wk = tid >> 3;            // 0..31
  const int wn = (tid & 7) * 8;       // 0..56

  // compute indices
  const int arow = wid * 16 + (lane & 15);
  const int col  = lane & 15;
  const int kb   = (lane < 16) ? 0 : 8;

  v8f acc[4] = {v8f{}, v8f{}, v8f{}, v8f{}};

  for (int k0 = 0; k0 < K; k0 += BK) {
    __syncthreads();
    {
      const float* ap = A + (size_t)(m0 + ar) * K + k0 + ac;
      const float4 a0 = *(const float4*)(ap);
      const float4 a1 = *(const float4*)(ap + 4);
      const float4 a2 = *(const float4*)(ap + 8);
      const float4 a3 = *(const float4*)(ap + 12);
      v8bf p0, p1;
      p0[0]=(bf16_t)a0.x; p0[1]=(bf16_t)a0.y; p0[2]=(bf16_t)a0.z; p0[3]=(bf16_t)a0.w;
      p0[4]=(bf16_t)a1.x; p0[5]=(bf16_t)a1.y; p0[6]=(bf16_t)a1.z; p0[7]=(bf16_t)a1.w;
      p1[0]=(bf16_t)a2.x; p1[1]=(bf16_t)a2.y; p1[2]=(bf16_t)a2.z; p1[3]=(bf16_t)a2.w;
      p1[4]=(bf16_t)a3.x; p1[5]=(bf16_t)a3.y; p1[6]=(bf16_t)a3.z; p1[7]=(bf16_t)a3.w;
      *(v8bf*)(As + ar * LSTR + ac)     = p0;
      *(v8bf*)(As + ar * LSTR + ac + 8) = p1;

      const float* wp = W + (size_t)(k0 + wk) * N + n0 + wn;
      const float4 w0 = *(const float4*)(wp);
      const float4 w1 = *(const float4*)(wp + 4);
      Wt[(wn + 0) * LSTR + wk] = (bf16_t)w0.x;
      Wt[(wn + 1) * LSTR + wk] = (bf16_t)w0.y;
      Wt[(wn + 2) * LSTR + wk] = (bf16_t)w0.z;
      Wt[(wn + 3) * LSTR + wk] = (bf16_t)w0.w;
      Wt[(wn + 4) * LSTR + wk] = (bf16_t)w1.x;
      Wt[(wn + 5) * LSTR + wk] = (bf16_t)w1.y;
      Wt[(wn + 6) * LSTR + wk] = (bf16_t)w1.z;
      Wt[(wn + 7) * LSTR + wk] = (bf16_t)w1.w;
    }
    __syncthreads();

    const v16bf afrag = load_frag_lds(As + arow * LSTR, kb);
#pragma unroll
    for (int ct = 0; ct < 4; ++ct) {
      const v16bf bfrag = load_frag_lds(Wt + (ct * 16 + col) * LSTR, kb);
      acc[ct] = wmma_bf16(afrag, bfrag, acc[ct]);
    }
  }

  const int rbase = (lane < 16) ? 0 : 8;
  const int gr = m0 + wid * 16 + rbase;
#pragma unroll
  for (int ct = 0; ct < 4; ++ct) {
    const int gc = n0 + ct * 16 + col;
    const float bv = bias[gc];
#pragma unroll
    for (int r = 0; r < 8; ++r) {
      float x = acc[ct][r] + bv;
      if (act == 1) x = gelu_exact(x);
      C[(size_t)(gr + r) * N + gc] = x;
    }
  }
}

// ---------------- Gate head: logg = log(clip(sigmoid(mu)*exp(-beta/(nu(alpha-1))))) ----
__global__ __launch_bounds__(256) void gate_finish_kernel(
    const float* __restrict__ hidden, const float* __restrict__ w2,
    const float* __restrict__ b2, float* __restrict__ logg, int M) {
  const int wid  = threadIdx.x >> 5;
  const int lane = threadIdx.x & 31;
  const int tok  = blockIdx.x * 8 + wid;
  if (tok >= M) return;
  const float* h = hidden + (size_t)tok * D_MODEL;
  float acc[4] = {0.f, 0.f, 0.f, 0.f};
  for (int c = lane; c < D_MODEL; c += 32) {
    const float hv = h[c];
    const float4 wv = *(const float4*)(w2 + c * 4);
    acc[0] += hv * wv.x; acc[1] += hv * wv.y;
    acc[2] += hv * wv.z; acc[3] += hv * wv.w;
  }
#pragma unroll
  for (int j = 0; j < 4; ++j)
#pragma unroll
    for (int off = 16; off > 0; off >>= 1)
      acc[j] += __shfl_xor(acc[j], off, 32);
  if (lane == 0) {
    const float mu    = acc[0] + b2[0];
    const float nu    = softplus_f(acc[1] + b2[1]) + 1e-4f;
    const float alpha = softplus_f(acc[2] + b2[2]) + 1.0f + 1e-4f;
    const float beta  = softplus_f(acc[3] + b2[3]) + 1e-4f;
    const float s2 = beta / (nu * (alpha - 1.0f));
    float g = (1.0f / (1.0f + expf(-mu))) * expf(-s2);
    g = fminf(fmaxf(g, 1e-3f), 1.0f);
    logg[tok] = logf(g);
  }
}

// ---------------- Flash attention, one wave per (b,h,16-row q tile) ----------------
__global__ __launch_bounds__(256) void attention_kernel(
    const float* __restrict__ Q, const float* __restrict__ K,
    const float* __restrict__ V, const float* __restrict__ logg,
    const int* __restrict__ mask, float* __restrict__ ctx,
    int Lq, int Lk) {
  __shared__ bf16_t Pst[8][16 * 32];  // per-wave P tile (row stride 32 bf16 = 64B)
  const int wid  = threadIdx.x >> 5;
  const int lane = threadIdx.x & 31;
  const int q0 = blockIdx.x * 128 + wid * 16;
  const int h  = blockIdx.y;
  const int b  = blockIdx.z;

  const float* Qb = Q + (size_t)(b * Lq + q0) * D_MODEL + h * DHEAD;
  const float* Kb = K + (size_t)b * Lk * D_MODEL + h * DHEAD;
  const float* Vb = V + (size_t)b * Lk * D_MODEL + h * DHEAD;
  const float* gb = logg + (size_t)b * Lk;
  const int*   mb = mask + (size_t)b * Lk;

  const v16bf qf0 = load_a_frag_g(Qb, D_MODEL);       // d 0..31
  const v16bf qf1 = load_a_frag_g(Qb + 32, D_MODEL);  // d 32..63

  const int col   = lane & 15;
  const int rbase = (lane < 16) ? 0 : 8;
  const int kb    = (lane < 16) ? 0 : 8;

  float m[8], l[8];
#pragma unroll
  for (int r = 0; r < 8; ++r) { m[r] = -1e30f; l[r] = 0.f; }
  v8f acc0 = {}, acc1 = {}, acc2 = {}, acc3 = {};

  const float qscale = 0.15625f;  // 1/(sqrt(64)*TEMP)
  const float invT   = 1.25f;     // 1/TEMP

  bf16_t* P = &Pst[wid][0];

  for (int kc = 0; kc < Lk; kc += 32) {
    // S = Q K^T for two 16-key tiles (K rows read as A-fragments == B of Q K^T)
    v8f s0 = {}, s1 = {};
    {
      const v16bf k00 = load_a_frag_g(Kb + (size_t)kc * D_MODEL, D_MODEL);
      const v16bf k01 = load_a_frag_g(Kb + (size_t)kc * D_MODEL + 32, D_MODEL);
      const v16bf k10 = load_a_frag_g(Kb + (size_t)(kc + 16) * D_MODEL, D_MODEL);
      const v16bf k11 = load_a_frag_g(Kb + (size_t)(kc + 16) * D_MODEL + 32, D_MODEL);
      s0 = wmma_bf16(qf0, k00, s0);
      s0 = wmma_bf16(qf1, k01, s0);
      s1 = wmma_bf16(qf0, k10, s1);
      s1 = wmma_bf16(qf1, k11, s1);
    }
    const int key0 = kc + col;
    const int key1 = kc + 16 + col;
    const float add0 = mb[key0] ? (gb[key0] * invT) : -1e30f;
    const float add1 = mb[key1] ? (gb[key1] * invT) : -1e30f;

    float f0[8], f1[8];
#pragma unroll
    for (int r = 0; r < 8; ++r) {
      f0[r] = s0[r] * qscale + add0;
      f1[r] = s1[r] * qscale + add1;
    }
    // online softmax (row stats reduced across the 16 lanes of each half-wave)
#pragma unroll
    for (int r = 0; r < 8; ++r) {
      float rmax = fmaxf(f0[r], f1[r]);
      rmax = fmaxf(rmax, __shfl_xor(rmax, 1, 32));
      rmax = fmaxf(rmax, __shfl_xor(rmax, 2, 32));
      rmax = fmaxf(rmax, __shfl_xor(rmax, 4, 32));
      rmax = fmaxf(rmax, __shfl_xor(rmax, 8, 32));
      const float nm  = fmaxf(m[r], rmax);
      const float fac = __expf(m[r] - nm);
      const float p0  = __expf(f0[r] - nm);
      const float p1  = __expf(f1[r] - nm);
      float rs = p0 + p1;
      rs += __shfl_xor(rs, 1, 32);
      rs += __shfl_xor(rs, 2, 32);
      rs += __shfl_xor(rs, 4, 32);
      rs += __shfl_xor(rs, 8, 32);
      l[r] = l[r] * fac + rs;
      m[r] = nm;
      acc0[r] = acc0[r] * fac;
      acc1[r] = acc1[r] * fac;
      acc2[r] = acc2[r] * fac;
      acc3[r] = acc3[r] * fac;
      P[(rbase + r) * 32 + col]      = (bf16_t)p0;
      P[(rbase + r) * 32 + 16 + col] = (bf16_t)p1;
    }
    asm volatile("s_wait_dscnt 0" ::: "memory");  // wave-private LDS RAW
    const v16bf pf = load_frag_lds(P + (lane & 15) * 32, kb);  // 2x ds_load_b128
    const float* vrow = Vb + (size_t)kc * D_MODEL;
    acc0 = wmma_bf16(pf, load_b_frag_g(vrow, D_MODEL), acc0);
    acc1 = wmma_bf16(pf, load_b_frag_g(vrow + 16, D_MODEL), acc1);
    acc2 = wmma_bf16(pf, load_b_frag_g(vrow + 32, D_MODEL), acc2);
    acc3 = wmma_bf16(pf, load_b_frag_g(vrow + 48, D_MODEL), acc3);
  }

  float* ob = ctx + (size_t)(b * Lq + q0) * D_MODEL + h * DHEAD;
#pragma unroll
  for (int r = 0; r < 8; ++r) {
    const float inv = 1.0f / l[r];
    ob[(size_t)(rbase + r) * D_MODEL + col]      = acc0[r] * inv;
    ob[(size_t)(rbase + r) * D_MODEL + 16 + col] = acc1[r] * inv;
    ob[(size_t)(rbase + r) * D_MODEL + 32 + col] = acc2[r] * inv;
    ob[(size_t)(rbase + r) * D_MODEL + 48 + col] = acc3[r] * inv;
  }
}

// ---------------- y = LayerNorm(x + r) * gamma + beta, wave per token ----------------
__global__ __launch_bounds__(256) void residual_ln_kernel(
    const float* __restrict__ x, const float* __restrict__ res,
    const float* __restrict__ gamma, const float* __restrict__ beta,
    float* __restrict__ y, int M) {
  const int wid  = threadIdx.x >> 5;
  const int lane = threadIdx.x & 31;
  const int tok  = blockIdx.x * 8 + wid;
  if (tok >= M) return;
  const float* xp = x + (size_t)tok * D_MODEL;
  const float* rp = res + (size_t)tok * D_MODEL;
  float v[24];
  float s = 0.f;
#pragma unroll
  for (int i = 0; i < 24; ++i) {
    const int c = lane + i * 32;
    v[i] = xp[c] + rp[c];
    s += v[i];
  }
#pragma unroll
  for (int off = 16; off > 0; off >>= 1) s += __shfl_xor(s, off, 32);
  const float mean = s * (1.0f / D_MODEL);
  float vs = 0.f;
#pragma unroll
  for (int i = 0; i < 24; ++i) { const float d = v[i] - mean; vs += d * d; }
#pragma unroll
  for (int off = 16; off > 0; off >>= 1) vs += __shfl_xor(vs, off, 32);
  const float rstd = rsqrtf(vs * (1.0f / D_MODEL) + LN_EPS);
  float* yp = y + (size_t)tok * D_MODEL;
#pragma unroll
  for (int i = 0; i < 24; ++i) {
    const int c = lane + i * 32;
    yp[c] = (v[i] - mean) * rstd * gamma[c] + beta[c];
  }
}

// ---------------- Host orchestration ----------------
struct ParamPtrs {
  const float *gd_w1, *gd_b1, *gd_w2, *gd_b2;
  const float *gp_w1, *gp_b1, *gp_w2, *gp_b2;
  const float *dp_wq, *dp_bq, *dp_wk, *dp_bk, *dp_wv, *dp_bv, *dp_wo, *dp_bo;
  const float *pd_wq, *pd_bq, *pd_wk, *pd_bk, *pd_wv, *pd_bv, *pd_wo, *pd_bo;
  const float *ld1_g, *ld1_b, *lp1_g, *lp1_b;
  const float *fd_w1, *fd_b1, *fd_w2, *fd_b2;
  const float *fp_w1, *fp_b1, *fp_w2, *fp_b2;
  const float *ld2_g, *ld2_b, *lp2_g, *lp2_b;
};

extern "C" void kernel_launch(void* const* d_in, const int* in_sizes, int n_in,
                              void* d_out, int out_size, void* d_ws, size_t ws_size,
                              hipStream_t stream) {
  (void)n_in; (void)out_size; (void)ws_size;
  const float* H_D    = (const float*)d_in[0];
  const float* H_P    = (const float*)d_in[1];
  const int*   mask_d = (const int*)d_in[2];
  const int*   mask_p = (const int*)d_in[3];

  auto F = [&](int i) { return (const float*)d_in[i]; };
  ParamPtrs p;
  if (in_sizes[4] > 100000) {
    // nested-dict insertion order
    p.gd_w1=F(4);  p.gd_b1=F(5);  p.gd_w2=F(6);  p.gd_b2=F(7);
    p.gp_w1=F(8);  p.gp_b1=F(9);  p.gp_w2=F(10); p.gp_b2=F(11);
    p.dp_wq=F(12); p.dp_bq=F(13); p.dp_wk=F(14); p.dp_bk=F(15);
    p.dp_wv=F(16); p.dp_bv=F(17); p.dp_wo=F(18); p.dp_bo=F(19);
    p.pd_wq=F(20); p.pd_bq=F(21); p.pd_wk=F(22); p.pd_bk=F(23);
    p.pd_wv=F(24); p.pd_bv=F(25); p.pd_wo=F(26); p.pd_bo=F(27);
    p.ld1_g=F(28); p.ld1_b=F(29); p.lp1_g=F(30); p.lp1_b=F(31);
    p.fd_w1=F(32); p.fd_b1=F(33); p.fd_w2=F(34); p.fd_b2=F(35);
    p.fp_w1=F(36); p.fp_b1=F(37); p.fp_w2=F(38); p.fp_b2=F(39);
    p.ld2_g=F(40); p.ld2_b=F(41); p.lp2_g=F(42); p.lp2_b=F(43);
  } else {
    // jax-pytree sorted-key order
    p.dp_bk=F(4);  p.dp_bo=F(5);  p.dp_bq=F(6);  p.dp_bv=F(7);
    p.dp_wk=F(8);  p.dp_wo=F(9);  p.dp_wq=F(10); p.dp_wv=F(11);
    p.pd_bk=F(12); p.pd_bo=F(13); p.pd_bq=F(14); p.pd_bv=F(15);
    p.pd_wk=F(16); p.pd_wo=F(17); p.pd_wq=F(18); p.pd_wv=F(19);
    p.fd_b1=F(20); p.fd_b2=F(21); p.fd_w1=F(22); p.fd_w2=F(23);
    p.fp_b1=F(24); p.fp_b2=F(25); p.fp_w1=F(26); p.fp_w2=F(27);
    p.gd_b1=F(28); p.gd_b2=F(29); p.gd_w1=F(30); p.gd_w2=F(31);
    p.gp_b1=F(32); p.gp_b2=F(33); p.gp_w1=F(34); p.gp_w2=F(35);
    p.ld1_b=F(36); p.ld1_g=F(37); p.ld2_b=F(38); p.ld2_g=F(39);
    p.lp1_b=F(40); p.lp1_g=F(41); p.lp2_b=F(42); p.lp2_g=F(43);
  }

  const int MD = BATCH * LQ_D;   // 8192 tokens (dialogue)
  const int MP = BATCH * LQ_P;   // 16384 tokens (passage)

  float* ws = (float*)d_ws;
  float* logg_d = ws;
  float* logg_p = ws + 8192;
  const size_t REG = (size_t)16384 * 768;
  float* B0 = ws + 32768;        // q_d    | Hd1
  float* B1 = B0 + REG;          // k_p    | attnout_d | ffn outs
  float* B2 = B1 + REG;          // v_p    | attnout_p
  float* B3 = B2 + REG;          // q_p    | Hp1
  float* B4 = B3 + REG;          // k_d    | ffn hidden base
  float* B5 = B4 + REG;          // v_d    | ffn hidden
  float* B6 = B5 + REG;          // gate hidden d | ctx_d
  float* B7 = B6 + REG;          // gate hidden p | ctx_p

  auto gemm = [&](const float* A, const float* W, const float* bias, float* C,
                  int M, int N, int K, int act) {
    dim3 g(N / BN, M / BM);
    gemm_bias_act_kernel<<<g, 256, 0, stream>>>(A, W, bias, C, M, N, K, act);
  };

  // 1) uncertainty gates -> log g
  gemm(H_D, p.gd_w1, p.gd_b1, B6, MD, D_MODEL, D_MODEL, 1);
  gate_finish_kernel<<<MD / 8, 256, 0, stream>>>(B6, p.gd_w2, p.gd_b2, logg_d, MD);
  gemm(H_P, p.gp_w1, p.gp_b1, B7, MP, D_MODEL, D_MODEL, 1);
  gate_finish_kernel<<<MP / 8, 256, 0, stream>>>(B7, p.gp_w2, p.gp_b2, logg_p, MP);

  // 2) Q/K/V projections
  gemm(H_D, p.dp_wq, p.dp_bq, B0, MD, D_MODEL, D_MODEL, 0);  // q_d
  gemm(H_P, p.dp_wk, p.dp_bk, B1, MP, D_MODEL, D_MODEL, 0);  // k_p
  gemm(H_P, p.dp_wv, p.dp_bv, B2, MP, D_MODEL, D_MODEL, 0);  // v_p
  gemm(H_P, p.pd_wq, p.pd_bq, B3, MP, D_MODEL, D_MODEL, 0);  // q_p
  gemm(H_D, p.pd_wk, p.pd_bk, B4, MD, D_MODEL, D_MODEL, 0);  // k_d
  gemm(H_D, p.pd_wv, p.pd_bv, B5, MD, D_MODEL, D_MODEL, 0);  // v_d

  // 3) gated cross attention (flash, per-head)
  dim3 gad(LQ_D / 128, NHEAD, BATCH);
  attention_kernel<<<gad, 256, 0, stream>>>(B0, B1, B2, logg_p, mask_p, B6, LQ_D, LQ_P);
  dim3 gap(LQ_P / 128, NHEAD, BATCH);
  attention_kernel<<<gap, 256, 0, stream>>>(B3, B4, B5, logg_d, mask_d, B7, LQ_P, LQ_D);

  // 4) output projections
  gemm(B6, p.dp_wo, p.dp_bo, B1, MD, D_MODEL, D_MODEL, 0);   // attnout_d
  gemm(B7, p.pd_wo, p.pd_bo, B2, MP, D_MODEL, D_MODEL, 0);   // attnout_p

  // 5) first residual + LN
  residual_ln_kernel<<<MD / 8, 256, 0, stream>>>(H_D, B1, p.ld1_g, p.ld1_b, B0, MD);  // Hd1
  residual_ln_kernel<<<MP / 8, 256, 0, stream>>>(H_P, B2, p.lp1_g, p.lp1_b, B3, MP);  // Hp1

  // 6) FFN(d) + LN2 -> d_out[0:MD*768]
  float* outHd = (float*)d_out;
  gemm(B0, p.fd_w1, p.fd_b1, B4, MD, DFF, D_MODEL, 1);       // hidden_d spans B4..B5
  gemm(B4, p.fd_w2, p.fd_b2, B1, MD, D_MODEL, DFF, 0);
  residual_ln_kernel<<<MD / 8, 256, 0, stream>>>(B0, B1, p.ld2_g, p.ld2_b, outHd, MD);

  // 7) FFN(p) + LN2 -> d_out[MD*768:]
  float* outHp = outHd + (size_t)MD * D_MODEL;
  gemm(B3, p.fp_w1, p.fp_b1, B4, MP, DFF, D_MODEL, 1);       // hidden_p spans B4..B7
  gemm(B4, p.fp_w2, p.fp_b2, B1, MP, D_MODEL, DFF, 0);
  residual_ln_kernel<<<MP / 8, 256, 0, stream>>>(B3, B1, p.lp2_g, p.lp2_b, outHp, MP);
}
